// MeanField_59304908423619
// MI455X (gfx1250) — matrix operand order repaired and checked by reading, compiled, not run on hardware
//
#include <hip/hip_runtime.h>
#include <hip/hip_bf16.h>

typedef __attribute__((ext_vector_type(16))) _Float16 v16h;
typedef __attribute__((ext_vector_type(8)))  float    v8f;

namespace {
constexpr int kB = 2, kC = 3, kH = 256, kW = 704, kD = 64;
constexpr int kHW = kH * kW;
constexpr int TN = 64;                        // pixels per tile (contiguous in W)
constexpr int QTS = kD + 16;                  // qt_t row stride: 80 halves = 160 B (32B multiple)
constexpr float kZeta   = 1e-3f;
constexpr float kInv2Z2 = 1.0f / (2.0f * kZeta * kZeta);
}

// ---------------------------------------------------------------------------
// Kernel 1: q0 = softmax(logits, axis=D), stored f16
// ---------------------------------------------------------------------------
__global__ __launch_bounds__(256) void softmax_init_kernel(
    const float* __restrict__ logits, _Float16* __restrict__ qout) {
  __shared__ float sl[kD][TN + 1];
  __shared__ float rinv[TN];
  const int tid = threadIdx.x;
  const int w0 = blockIdx.x * TN;
  const int h  = blockIdx.y;
  const int b  = blockIdx.z;
  const size_t base = (size_t)(b * kD) * kHW + (size_t)h * kW + w0;

  for (int idx = tid; idx < kD * TN; idx += 256) {
    int d = idx >> 6, j = idx & 63;
    sl[d][j] = logits[base + (size_t)d * kHW + j];
  }
  __syncthreads();
  if (tid < TN) {
    int j = tid;
    float mx = -3.0e38f;
    #pragma unroll 8
    for (int d = 0; d < kD; ++d) mx = fmaxf(mx, sl[d][j]);
    float sum = 0.0f;
    #pragma unroll 8
    for (int d = 0; d < kD; ++d) {
      float e = expf(sl[d][j] - mx);
      sl[d][j] = e;
      sum += e;
    }
    rinv[j] = 1.0f / sum;
  }
  __syncthreads();
  for (int idx = tid; idx < kD * TN; idx += 256) {
    int d = idx >> 6, j = idx & 63;
    qout[base + (size_t)d * kHW + j] = (_Float16)(sl[d][j] * rinv[j]);
  }
}

// ---------------------------------------------------------------------------
// Kernel 2: one mean-field iteration, fused:
//   kern = exp-affinity(color);  q_tilde = sum_k kern_k * q_in(neighbor_k)
//   q_hat = ws * (miu @ q_tilde)   [WMMA f16 hi/lo split, f32 accumulate]
//   q_out = softmax(logits - q_hat)
// ---------------------------------------------------------------------------
template <bool LAST>
__global__ __launch_bounds__(256) void crf_iter_kernel(
    const _Float16* __restrict__ qin, const float* __restrict__ color,
    const float* __restrict__ logits, const float* __restrict__ wsp,
    _Float16* __restrict__ qout_h, float* __restrict__ qout_f) {
  __shared__ float    cstage[kC][3][TN + 2];
  __shared__ float    kern[9][TN];
  __shared__ _Float16 qstage[kD][3][TN + 2];
  // q_tilde stored TRANSPOSED: [pixel j][channel e], 32B-aligned rows so a
  // wave's B-fragment (16 consecutive e at fixed j) is one v16h LDS load.
  __shared__ alignas(32) _Float16 qt_hi_t[TN][QTS];
  __shared__ alignas(32) _Float16 qt_lo_t[TN][QTS];
  __shared__ float    s_lds[kD][TN + 1];
  __shared__ float    rinv[TN];

  const int tid = threadIdx.x;
  const int w0 = blockIdx.x * TN;
  const int h  = blockIdx.y;
  const int b  = blockIdx.z;
  const float wsv = wsp[0];

  // ---- Phase 0: stage color halo (f32) and q_in halo (f16); zero padding ----
  for (int idx = tid; idx < kC * 3 * (TN + 2); idx += 256) {
    int c   = idx / (3 * (TN + 2));
    int rem = idx - c * 3 * (TN + 2);
    int r   = rem / (TN + 2);
    int jj  = rem - r * (TN + 2);
    int hh = h + r - 1;
    int ww = w0 + jj - 1;
    float v = 0.0f;
    if (hh >= 0 && hh < kH && ww >= 0 && ww < kW)
      v = color[(size_t)(b * kC + c) * kHW + (size_t)hh * kW + ww];
    cstage[c][r][jj] = v;
  }
  for (int idx = tid; idx < kD * 3 * (TN + 2); idx += 256) {
    int d   = idx / (3 * (TN + 2));
    int rem = idx - d * 3 * (TN + 2);
    int r   = rem / (TN + 2);
    int jj  = rem - r * (TN + 2);
    int hh = h + r - 1;
    int ww = w0 + jj - 1;
    _Float16 v = (_Float16)0.0f;
    if (hh >= 0 && hh < kH && ww >= 0 && ww < kW)
      v = qin[(size_t)(b * kD + d) * kHW + (size_t)hh * kW + ww];
    qstage[d][r][jj] = v;
  }
  __syncthreads();

  // ---- Phase 0b: gaussian affinity kernel (9 taps per pixel) ----
  for (int idx = tid; idx < 9 * TN; idx += 256) {
    int k = idx >> 6;          // 0..8
    int j = idx & 63;
    int dy = k / 3, dx = k - dy * 3;
    float ssum = 0.0f;
    #pragma unroll
    for (int c = 0; c < kC; ++c) {
      float dlt = cstage[c][dy][j + dx] - cstage[c][1][j + 1];
      ssum += dlt * dlt;
    }
    kern[k][j] = expf(-ssum * kInv2Z2);
  }
  __syncthreads();

  // ---- Phase 1: q_tilde (ws folded in later), split f16 hi + lo, transposed ----
  {
    int j  = tid & 63;
    int dg = tid >> 6;
    for (int d = dg; d < kD; d += 4) {
      float acc = 0.0f;
      #pragma unroll
      for (int k = 0; k < 9; ++k) {
        int dy = k / 3, dx = k - dy * 3;
        acc += kern[k][j] * (float)qstage[d][dy][j + dx];
      }
      _Float16 hi = (_Float16)acc;
      qt_hi_t[j][d] = hi;
      qt_lo_t[j][d] = (_Float16)(acc - (float)hi);
    }
  }
  __syncthreads();

  // ---- Phase 2: q_hat = miu @ q_tilde via V_WMMA_F32_16X16X32_F16 ----
  // 8 waves: wave w -> pixel block nb = w&3 (16 px), D-row blocks db0, db0+1.
  const int wave = tid >> 5;
  const int lane = tid & 31;
  const int g = lane >> 4;
  const int m = lane & 15;
  const int nb  = wave & 3;
  const int db0 = (wave >> 2) * 2;
  const int jn  = nb * 16 + m;

  // Hoist the 4 constant miu A-fragments (A: M = m, K = i+16*g).
  v16h a_frag[2][2];
  #pragma unroll
  for (int kc = 0; kc < 2; ++kc) {
    #pragma unroll
    for (int dd = 0; dd < 2; ++dd) {
      int db = db0 + dd;
      #pragma unroll
      for (int i = 0; i < 16; ++i) {
        int kk = i + g * 16;
        int e = kc * 32 + kk;
        int d = db * 16 + m;
        a_frag[kc][dd][i] = (_Float16)fabsf((float)(d - e));  // miu = |d-e|, exact in f16
      }
    }
  }

  v8f acc[2] = {};
  #pragma unroll
  for (int kc = 0; kc < 2; ++kc) {
    // B: K = kc*32 + 16*g + i, N = m -> contiguous halves in transposed qt.
    const v16h bh = *(const v16h*)&qt_hi_t[jn][kc * 32 + g * 16];
    const v16h bl = *(const v16h*)&qt_lo_t[jn][kc * 32 + g * 16];
    #pragma unroll
    for (int dd = 0; dd < 2; ++dd) {
      acc[dd] = __builtin_amdgcn_wmma_f32_16x16x32_f16(
          false, a_frag[kc][dd], false, bh, (short)0, acc[dd], false, false);
      acc[dd] = __builtin_amdgcn_wmma_f32_16x16x32_f16(
          false, a_frag[kc][dd], false, bl, (short)0, acc[dd], false, false);
    }
  }

  // ---- Phase 3: s = logits - ws * q_hat  (C/D layout: M = r + 8*g) ----
  {
    const size_t base = (size_t)(b * kD) * kHW + (size_t)h * kW + w0;
    #pragma unroll
    for (int dd = 0; dd < 2; ++dd) {
      int db = db0 + dd;
      #pragma unroll
      for (int r = 0; r < 8; ++r) {
        int d = db * 16 + g * 8 + r;
        float lg = logits[base + (size_t)d * kHW + jn];
        s_lds[d][jn] = lg - wsv * acc[dd][r];
      }
    }
  }
  __syncthreads();

  // ---- Phase 4: per-pixel softmax over D ----
  if (tid < TN) {
    int j = tid;
    float mx = -3.0e38f;
    #pragma unroll 8
    for (int d = 0; d < kD; ++d) mx = fmaxf(mx, s_lds[d][j]);
    float sum = 0.0f;
    #pragma unroll 8
    for (int d = 0; d < kD; ++d) {
      float e = expf(s_lds[d][j] - mx);
      s_lds[d][j] = e;
      sum += e;
    }
    rinv[j] = 1.0f / sum;
  }
  __syncthreads();

  // ---- Phase 5: coalesced writeback (f16 intermediate or f32 final) ----
  {
    const size_t base = (size_t)(b * kD) * kHW + (size_t)h * kW + w0;
    for (int idx = tid; idx < kD * TN; idx += 256) {
      int d = idx >> 6, j = idx & 63;
      float v = s_lds[d][j] * rinv[j];
      if (LAST) qout_f[base + (size_t)d * kHW + j] = v;
      else      qout_h[base + (size_t)d * kHW + j] = (_Float16)v;
    }
  }
}

// ---------------------------------------------------------------------------
extern "C" void kernel_launch(void* const* d_in, const int* in_sizes, int n_in,
                              void* d_out, int out_size, void* d_ws, size_t ws_size,
                              hipStream_t stream) {
  const float* color  = (const float*)d_in[0];
  // d_in[1] = feats: unused by the reference forward pass
  const float* logits = (const float*)d_in[2];
  const float* wsp    = (const float*)d_in[3];
  float* out = (float*)d_out;

  // scratch: two f16 q buffers (46 MB each)
  _Float16* qA = (_Float16*)d_ws;
  _Float16* qB = qA + (size_t)kB * kD * kHW;

  dim3 grid(kW / TN, kH, kB);   // 11 x 256 x 2 tiles, 64 pixels each
  softmax_init_kernel<<<grid, 256, 0, stream>>>(logits, qA);
  crf_iter_kernel<false><<<grid, 256, 0, stream>>>(qA, color, logits, wsp, qB, nullptr);
  crf_iter_kernel<true ><<<grid, 256, 0, stream>>>(qB, color, logits, wsp, nullptr, out);

  (void)in_sizes; (void)n_in; (void)out_size; (void)ws_size;
}
